// VarianceAdaptor_33672543601371
// MI455X (gfx1250) — matrix operand order repaired and checked by reading, not compiled, not run on hardware
//
#include <hip/hip_runtime.h>
#include <hip/hip_bf16.h>

// ---------------------------------------------------------------------------
// Problem constants (match reference)
// ---------------------------------------------------------------------------
#define BATCH   64
#define SEQ     512
#define CH      256     // H == F == 256
#define KTAPS   3
#define TMAX    2048
#define NBINS   256

#define TILE_M    64
#define AS_ROWS   66            // 64 + 2 halo
#define AS_STRIDE 264           // halfs; 528B row stride -> bank rotation
#define HS_STRIDE 260           // floats

typedef __bf16 bf16_t;
typedef __attribute__((ext_vector_type(16))) __bf16 v16bf;
typedef __attribute__((ext_vector_type(8)))  __bf16 v8bf;
typedef __attribute__((ext_vector_type(8)))  float  v8f;
typedef __attribute__((ext_vector_type(4)))  int    v4i;

#define AS1 __attribute__((address_space(1)))
#define AS3 __attribute__((address_space(3)))

#if __has_builtin(__builtin_amdgcn_global_load_async_to_lds_b128)
#define HAVE_ASYNC_LDS 1
#endif

union AFrag { v16bf f; v8bf h[2]; };

__device__ __forceinline__ int bin_index(float v) {
    // searchsorted(linspace(-2,2,255), v, side='left')
    float t = ceilf((v + 2.0f) * 63.5f);
    int i = (int)t;
    i = i < 0 ? 0 : i;
    i = i > (NBINS - 1) ? (NBINS - 1) : i;
    return i;
}

// ---------------------------------------------------------------------------
// Conversions
// ---------------------------------------------------------------------------
__global__ void cvt_x_kernel(const float* __restrict__ in, bf16_t* __restrict__ out, int n) {
    int i = blockIdx.x * 256 + threadIdx.x;
    if (i < n) out[i] = (bf16_t)in[i];
}

// (K,H,F) f32  ->  (K,F,H) bf16  (transposed so WMMA B-fragments are contiguous)
__global__ void cvt_w_kernel(const float* __restrict__ src, bf16_t* __restrict__ dst) {
    int i = blockIdx.x * 256 + threadIdx.x;
    if (i < KTAPS * CH * CH) {
        int k   = i / (CH * CH);
        int rem = i % (CH * CH);
        int c   = rem / CH;       // input channel
        int f   = rem % CH;       // output channel
        dst[((size_t)k * CH + f) * CH + c] = (bf16_t)src[i];
    }
}

// xpeb[b,s,:] = bf16( x[b,s,:] + p_table[bucket(pitch[b,s])] )
__global__ void add_emb_kernel(const float* __restrict__ x,
                               const float* __restrict__ tgt,
                               const float* __restrict__ table,
                               bf16_t* __restrict__ out) {
    int bs = blockIdx.x;
    int c  = threadIdx.x;
    int idx = bin_index(tgt[bs]);
    out[(size_t)bs * CH + c] = (bf16_t)(x[(size_t)bs * CH + c] + table[(size_t)idx * CH + c]);
}

// ---------------------------------------------------------------------------
// Fused conv1d(K=3) -> bias -> ReLU -> LayerNorm   (WMMA bf16, f32 accum)
//   STAGE2: additionally fuse 256->1 projection + mask, write prediction.
// Grid: BATCH * (SEQ/TILE_M) blocks, 256 threads (8 waves).
// ---------------------------------------------------------------------------
template <int STAGE2>
__global__ __launch_bounds__(256, 2)
void conv_ln_kernel(const bf16_t* __restrict__ X,     // (B,S,CH) bf16
                    const bf16_t* __restrict__ Wt,    // (K,F,CH) bf16 transposed
                    const float*  __restrict__ bias,
                    const float*  __restrict__ gamma,
                    const float*  __restrict__ beta,
                    const float*  __restrict__ lw,    // (CH) stage2 only
                    const float*  __restrict__ lb,    // (1)  stage2 only
                    const unsigned char* __restrict__ mask, // (B,S) stage2 only
                    bf16_t* __restrict__ Hout,        // (B,S,CH) stage1 output
                    float*  __restrict__ Pred)        // (B,S)    stage2 output
{
    __shared__ bf16_t As[AS_ROWS * AS_STRIDE];
    __shared__ float  Hs[TILE_M * HS_STRIDE];
    __shared__ float  row_mu[TILE_M];
    __shared__ float  row_inv[TILE_M];

    const int tid = threadIdx.x;
    const int b   = blockIdx.x >> 3;            // SEQ / TILE_M == 8
    const int s0  = (blockIdx.x & 7) * TILE_M;

    // ---- stage A tile (66 rows x 256 ch) into LDS, zero halo at edges ----
    for (int i = tid; i < AS_ROWS * (CH / 8); i += 256) {
        int row = i >> 5;                // 32 chunks of 8 halfs per row
        int ch  = (i & 31) << 3;
        int s   = s0 - 1 + row;
        bf16_t* lp = &As[row * AS_STRIDE + ch];
        if (s >= 0 && s < SEQ) {
#ifdef HAVE_ASYNC_LDS
            // CDNA5 async direct-to-LDS copy (no VGPR round trip), ASYNCcnt-tracked
            __builtin_amdgcn_global_load_async_to_lds_b128(
                (AS1 v4i*)(X + ((size_t)b * SEQ + s) * CH + ch),
                (AS3 v4i*)lp, 0, 0);
#else
            *(v8bf*)lp = *(const v8bf*)(X + ((size_t)b * SEQ + s) * CH + ch);
#endif
        } else {
            v8bf z;
            #pragma unroll
            for (int j = 0; j < 8; ++j) z[j] = (__bf16)0.0f;
            *(v8bf*)lp = z;
        }
    }
#ifdef HAVE_ASYNC_LDS
    asm volatile("s_wait_asynccnt 0x0" ::: "memory");
#endif
    __syncthreads();

    const int wave = tid >> 5;
    const int lane = tid & 31;
    const int mset = wave & 1;        // 2 M-tiles per wave
    const int nset = wave >> 1;       // 4 N-tiles per wave
    const int lm   = lane & 15;
    const int lk   = lane >> 4;

    v8f acc[2][4];
    #pragma unroll
    for (int mi = 0; mi < 2; ++mi)
        #pragma unroll
        for (int ni = 0; ni < 4; ++ni)
            #pragma unroll
            for (int r = 0; r < 8; ++r) acc[mi][ni][r] = 0.0f;

    // ---- K loop: 3 taps x 8 chunks of K=32 ----
    for (int kk = 0; kk < KTAPS * (CH / 32); ++kk) {
        const int tap = kk >> 3;
        const int kc  = (kk & 7) << 5;

        // prefetch next K-chunk's weight columns (overlaps L2 latency w/ WMMAs)
        if (kk + 1 < KTAPS * (CH / 32)) {
            const int tap2 = (kk + 1) >> 3;
            const int kc2  = ((kk + 1) & 7) << 5;
            #pragma unroll
            for (int ni = 0; ni < 4; ++ni) {
                int n = nset * 64 + ni * 16 + lm;
                __builtin_prefetch(Wt + ((size_t)tap2 * CH + n) * CH + kc2 + lk * 16, 0, 1);
            }
        }

        AFrag a[2];
        #pragma unroll
        for (int mi = 0; mi < 2; ++mi) {
            // A layout: lanes 0-15 rows M, halfs K = {kc+lk*8 .. +7, kc+16+lk*8 .. +7}
            int row = (mset * 2 + mi) * 16 + lm + tap;      // 0..65
            const bf16_t* p = &As[row * AS_STRIDE + kc + lk * 8];
            a[mi].h[0] = *(const v8bf*)p;
            a[mi].h[1] = *(const v8bf*)(p + 16);
        }

        AFrag bfr[4];
        #pragma unroll
        for (int ni = 0; ni < 4; ++ni) {
            // B layout: lane -> column N, halfs K = {kc+lk*8 .. +7, kc+16+lk*8 .. +7}
            int n = nset * 64 + ni * 16 + lm;
            const bf16_t* p = Wt + ((size_t)tap * CH + n) * CH + kc + lk * 8;
            bfr[ni].h[0] = *(const v8bf*)p;
            bfr[ni].h[1] = *(const v8bf*)(p + 16);
        }

        #pragma unroll
        for (int mi = 0; mi < 2; ++mi)
            #pragma unroll
            for (int ni = 0; ni < 4; ++ni)
                acc[mi][ni] = __builtin_amdgcn_wmma_f32_16x16x32_bf16(
                    false, a[mi].f, false, bfr[ni].f,
                    (short)0, acc[mi][ni], false, false);
    }

    // ---- spill accumulators to LDS (C/D layout: VGPR r -> M = r + 8*lk) ----
    #pragma unroll
    for (int mi = 0; mi < 2; ++mi) {
        int mrow = (mset * 2 + mi) * 16 + lk * 8;
        #pragma unroll
        for (int ni = 0; ni < 4; ++ni) {
            int col = nset * 64 + ni * 16 + lm;
            #pragma unroll
            for (int r = 0; r < 8; ++r)
                Hs[(mrow + r) * HS_STRIDE + col] = acc[mi][ni][r];
        }
    }
    __syncthreads();

    // ---- bias + ReLU + LayerNorm stats (one row per thread, 64 rows) ----
    if (tid < TILE_M) {
        float* hrow = &Hs[tid * HS_STRIDE];
        float sum = 0.0f, sq = 0.0f;
        #pragma unroll 4
        for (int c = 0; c < CH; ++c) {
            float v = hrow[c] + bias[c];
            v = v > 0.0f ? v : 0.0f;
            hrow[c] = v;
            sum += v;
            sq  += v * v;
        }
        float mu  = sum * (1.0f / CH);
        float var = sq * (1.0f / CH) - mu * mu;
        float inv = __frsqrt_rn(var + 1e-5f);

        if (STAGE2) {
            float dot = 0.0f;
            #pragma unroll 4
            for (int c = 0; c < CH; ++c) {
                float ln = (hrow[c] - mu) * inv * gamma[c] + beta[c];
                dot += ln * lw[c];
            }
            int s = s0 + tid;
            float outv = dot + lb[0];
            if (mask[(size_t)b * SEQ + s]) outv = 0.0f;
            Pred[(size_t)b * SEQ + s] = outv;
        } else {
            row_mu[tid]  = mu;
            row_inv[tid] = inv;
        }
    }

    if (!STAGE2) {
        __syncthreads();
        // ---- column-parallel, fully coalesced bf16 output writes ----
        const int col = tid;
        const float g  = gamma[col];
        const float be = beta[col];
        #pragma unroll 4
        for (int r = 0; r < TILE_M; ++r) {
            float ln = (Hs[r * HS_STRIDE + col] - row_mu[r]) * row_inv[r] * g + be;
            Hout[((size_t)b * SEQ + s0 + r) * CH + col] = (bf16_t)ln;
        }
    }
}

// ---------------------------------------------------------------------------
// Length regulator pieces
// ---------------------------------------------------------------------------
__global__ void cumsum_kernel(const int* __restrict__ dur,
                              int* __restrict__ cum,
                              float* __restrict__ mel_len_out) {
    int b = threadIdx.x;
    if (b < BATCH) {
        int run = 0;
        for (int s = 0; s < SEQ; ++s) {
            run += dur[b * SEQ + s];
            cum[b * SEQ + s] = run;
        }
        mel_len_out[b] = (float)run;
    }
}

__global__ void dur_to_float_kernel(const int* __restrict__ dur, float* __restrict__ out, int n) {
    int i = blockIdx.x * 256 + threadIdx.x;
    if (i < n) out[i] = (float)dur[i];
}

// mel[b,t,:] = (t < len) ? x[b,idx,:] + p_tab[pb(idx)] + e_tab[eb(idx)] : 0
__global__ void mel_kernel(const float* __restrict__ x,
                           const float* __restrict__ pitch,
                           const float* __restrict__ energy,
                           const float* __restrict__ ptab,
                           const float* __restrict__ etab,
                           const int*   __restrict__ cum,
                           float* __restrict__ mel) {
    int t = blockIdx.x & (TMAX - 1);
    int b = blockIdx.x >> 11;          // TMAX == 2048
    int c = threadIdx.x;
    const int* crow = cum + b * SEQ;
    int len = crow[SEQ - 1];
    size_t o = ((size_t)b * TMAX + t) * CH + c;
    if (t >= len) { mel[o] = 0.0f; return; }
    // searchsorted(cum, t, 'right') : first idx with cum[idx] > t
    int lo = 0, hi = SEQ;
    while (lo < hi) {
        int mid = (lo + hi) >> 1;
        if (crow[mid] <= t) lo = mid + 1; else hi = mid;
    }
    int idx = lo > (SEQ - 1) ? (SEQ - 1) : lo;
    int pi = bin_index(pitch[b * SEQ + idx]);
    int ei = bin_index(energy[b * SEQ + idx]);
    mel[o] = x[((size_t)b * SEQ + idx) * CH + c]
           + ptab[(size_t)pi * CH + c]
           + etab[(size_t)ei * CH + c];
}

// ---------------------------------------------------------------------------
// Launcher
// ---------------------------------------------------------------------------
extern "C" void kernel_launch(void* const* d_in, const int* in_sizes, int n_in,
                              void* d_out, int out_size, void* d_ws, size_t ws_size,
                              hipStream_t stream) {
    (void)in_sizes; (void)n_in; (void)out_size; (void)ws_size;

    const float* x        = (const float*)d_in[0];
    const float* pitch    = (const float*)d_in[1];
    const float* energy   = (const float*)d_in[2];
    const unsigned char* src_mask = (const unsigned char*)d_in[3];
    const int*   duration = (const int*)d_in[4];
    // d_in[5] = max_len scalar (compile-time TMAX)

    // per-predictor params: base 6 + 10*p : w1,b1,g1,be1,w2,b2,g2,be2,lw,lb
    auto P = [&](int pred, int j) -> const float* { return (const float*)d_in[6 + pred * 10 + j]; };
    const float* p_table = (const float*)d_in[36];
    const float* e_table = (const float*)d_in[37];

    // ---- workspace layout ----
    char* ws = (char*)d_ws;
    const size_t xb_bytes = (size_t)BATCH * SEQ * CH * sizeof(bf16_t);   // 16 MB
    bf16_t* xb    = (bf16_t*)(ws);
    bf16_t* xpeb  = (bf16_t*)(ws + xb_bytes);
    bf16_t* h1    = (bf16_t*)(ws + 2 * xb_bytes);
    const size_t w_bytes = (size_t)KTAPS * CH * CH * sizeof(bf16_t);     // 384 KB
    bf16_t* wt[6];
    for (int i = 0; i < 6; ++i) wt[i] = (bf16_t*)(ws + 3 * xb_bytes + i * w_bytes);
    int* cum = (int*)(ws + 3 * xb_bytes + 6 * w_bytes);

    // ---- output layout (all f32, concatenated) ----
    float* out      = (float*)d_out;
    float* mel      = out;                                               // B*T*CH
    float* p_pred   = out + (size_t)BATCH * TMAX * CH;
    float* e_pred   = p_pred + (size_t)BATCH * SEQ;
    float* logd     = e_pred + (size_t)BATCH * SEQ;
    float* d_round  = logd   + (size_t)BATCH * SEQ;
    float* mel_len  = d_round + (size_t)BATCH * SEQ;

    const int nX = BATCH * SEQ * CH;
    cvt_x_kernel<<<(nX + 255) / 256, 256, 0, stream>>>(x, xb, nX);

    const int nW = KTAPS * CH * CH;
    // weights: pred 0 (d): j=0 -> w1, j=4 -> w2 ; pred 1 (p), pred 2 (e)
    for (int pred = 0; pred < 3; ++pred) {
        cvt_w_kernel<<<(nW + 255) / 256, 256, 0, stream>>>(P(pred, 0), wt[pred * 2 + 0]);
        cvt_w_kernel<<<(nW + 255) / 256, 256, 0, stream>>>(P(pred, 4), wt[pred * 2 + 1]);
    }

    const dim3 cgrid(BATCH * (SEQ / TILE_M));
    const dim3 cblk(256);

    // duration predictor
    conv_ln_kernel<0><<<cgrid, cblk, 0, stream>>>(xb, wt[0], P(0,1), P(0,2), P(0,3),
                                                  nullptr, nullptr, nullptr, h1, nullptr);
    conv_ln_kernel<1><<<cgrid, cblk, 0, stream>>>(h1, wt[1], P(0,5), P(0,6), P(0,7),
                                                  P(0,8), P(0,9), src_mask, nullptr, logd);
    // pitch predictor
    conv_ln_kernel<0><<<cgrid, cblk, 0, stream>>>(xb, wt[2], P(1,1), P(1,2), P(1,3),
                                                  nullptr, nullptr, nullptr, h1, nullptr);
    conv_ln_kernel<1><<<cgrid, cblk, 0, stream>>>(h1, wt[3], P(1,5), P(1,6), P(1,7),
                                                  P(1,8), P(1,9), src_mask, nullptr, p_pred);
    // x + pitch embedding (bf16) for energy predictor
    add_emb_kernel<<<BATCH * SEQ, 256, 0, stream>>>(x, pitch, p_table, xpeb);
    // energy predictor
    conv_ln_kernel<0><<<cgrid, cblk, 0, stream>>>(xpeb, wt[4], P(2,1), P(2,2), P(2,3),
                                                  nullptr, nullptr, nullptr, h1, nullptr);
    conv_ln_kernel<1><<<cgrid, cblk, 0, stream>>>(h1, wt[5], P(2,5), P(2,6), P(2,7),
                                                  P(2,8), P(2,9), src_mask, nullptr, e_pred);

    // length regulator
    cumsum_kernel<<<1, 64, 0, stream>>>(duration, cum, mel_len);
    const int nD = BATCH * SEQ;
    dur_to_float_kernel<<<(nD + 255) / 256, 256, 0, stream>>>(duration, d_round, nD);
    mel_kernel<<<BATCH * TMAX, 256, 0, stream>>>(x, pitch, energy, p_table, e_table, cum, mel);
}
